// RNNAgent_New_65489661329787
// MI455X (gfx1250) — compile-verified, matrix operand order checked by reading, not run on hardware
//
#include <hip/hip_runtime.h>
#include <hip/hip_bf16.h>

#define BS_  4096
#define IN_  256
#define H_   256
#define G_   64
#define E_   256
#define A_   30
#define NW_  (IN_ * H_)
#define CH_  4      // i-chunks for the hyper kernel
#define ICH_ 64     // i per chunk
#define IPW_ 8      // i per wave (8 waves per group)

typedef __bf16 bf16_t;
typedef __attribute__((ext_vector_type(16))) __bf16 v16bf;
typedef __attribute__((ext_vector_type(8)))  __bf16 v8bf;
typedef __attribute__((ext_vector_type(8)))  float  v8f;

__device__ __forceinline__ unsigned short f2bf(float f) {
  unsigned u = __builtin_bit_cast(unsigned, f);
  u += 0x7FFFu + ((u >> 16) & 1u);           // round-to-nearest-even
  return (unsigned short)(u >> 16);
}

// Branch-free transcendentals on the TRANS32 pipe (no libm branchy paths).
__device__ __forceinline__ float fast_tanh(float x) {
#if __has_builtin(__builtin_amdgcn_tanhf)
  return __builtin_amdgcn_tanhf(x);          // v_tanh_f32 (gfx1250 native)
#else
  // tanh(x) = 1 - 2/(exp(2x)+1); saturates correctly via inf/0 of exp2.
  float e2 = __builtin_amdgcn_exp2f(x * 2.8853900817779268f);
  return 1.f - 2.f * __builtin_amdgcn_rcpf(e2 + 1.f);
#endif
}

__device__ __forceinline__ float fast_sigmoid(float x) {
  float e = __builtin_amdgcn_exp2f(-x * 1.4426950408889634f);
  return __builtin_amdgcn_rcpf(1.f + e);
}

// Load one 16-element bf16 WMMA fragment for this lane.
// 16-bit operand layout: lanes 0-15 hold K = k0+{0..7, 16..23},
// lanes 16-31 hold K = k0+{8..15, 24..31} (row/col given by lane&15).
__device__ __forceinline__ v16bf load_frag(const bf16_t* rowp, int k0, int half) {
  union { v16bf v; v8bf h[2]; } u;
  u.h[0] = *(const v8bf*)(rowp + k0 + half * 8);
  u.h[1] = *(const v8bf*)(rowp + k0 + 16 + half * 8);
  return u.v;
}

__device__ __forceinline__ float wave_sum(float v) {
#pragma unroll
  for (int m = 16; m >= 1; m >>= 1) v += __shfl_xor(v, m, 32);
  return v;
}

// ---------------- elementwise / small kernels ----------------

__global__ __launch_bounds__(256) void k_layernorm(
    const float* __restrict__ x, const float* __restrict__ g,
    const float* __restrict__ b, float* __restrict__ y, int rows) {
  int wid = threadIdx.x >> 5, lane = threadIdx.x & 31;
  int row = blockIdx.x * 8 + wid;
  if (row >= rows) return;
  const float* xr = x + (size_t)row * 256;
  float v[8]; float s = 0.f;
#pragma unroll
  for (int j = 0; j < 8; ++j) { v[j] = xr[lane + 32 * j]; s += v[j]; }
  float mu = wave_sum(s) * (1.f / 256.f);
  float q = 0.f;
#pragma unroll
  for (int j = 0; j < 8; ++j) { float d = v[j] - mu; q += d * d; }
  float rstd = rsqrtf(wave_sum(q) * (1.f / 256.f) + 1e-5f);
  float* yr = y + (size_t)row * 256;
#pragma unroll
  for (int j = 0; j < 8; ++j) {
    int c = lane + 32 * j;
    yr[c] = (v[j] - mu) * rstd * g[c] + b[c];
  }
}

__global__ __launch_bounds__(256) void k_ln_relu_bf(
    const float* __restrict__ x, const float* __restrict__ g,
    const float* __restrict__ b, unsigned short* __restrict__ ybf, int rows) {
  int wid = threadIdx.x >> 5, lane = threadIdx.x & 31;
  int row = blockIdx.x * 8 + wid;
  if (row >= rows) return;
  const float* xr = x + (size_t)row * 256;
  float v[8]; float s = 0.f;
#pragma unroll
  for (int j = 0; j < 8; ++j) { v[j] = xr[lane + 32 * j]; s += v[j]; }
  float mu = wave_sum(s) * (1.f / 256.f);
  float q = 0.f;
#pragma unroll
  for (int j = 0; j < 8; ++j) { float d = v[j] - mu; q += d * d; }
  float rstd = rsqrtf(wave_sum(q) * (1.f / 256.f) + 1e-5f);
  unsigned short* yr = ybf + (size_t)row * 256;
#pragma unroll
  for (int j = 0; j < 8; ++j) {
    int c = lane + 32 * j;
    float o = (v[j] - mu) * rstd * g[c] + b[c];
    yr[c] = f2bf(fmaxf(o, 0.f));
  }
}

__global__ __launch_bounds__(256) void k_h1(
    const float* __restrict__ goal, const float* __restrict__ w1,
    const float* __restrict__ b1, float* __restrict__ h1f,
    unsigned short* __restrict__ h1bf) {
  int idx = blockIdx.x * 256 + threadIdx.x;  // b*256 + e
  int b = idx >> 8, e = idx & 255;
  const float4* gr = (const float4*)(goal + (size_t)b * G_);
  const float4* wr = (const float4*)(w1 + (size_t)e * G_);
  float s = b1[e];
#pragma unroll
  for (int j = 0; j < G_ / 4; ++j) {
    float4 a = gr[j], w = wr[j];
    s += a.x * w.x + a.y * w.y + a.z * w.z + a.w * w.w;
  }
  s = fmaxf(s, 0.f);
  h1f[idx] = s;
  h1bf[idx] = f2bf(s);
}

__global__ __launch_bounds__(256) void k_f32_to_bf16(
    const float* __restrict__ x, unsigned short* __restrict__ y, size_t n) {
  size_t i = (size_t)blockIdx.x * 256 + threadIdx.x;
  if (i < n) y[i] = f2bf(x[i]);
}

// ---------------- fused hypernet GEMM + tanh + per-sample contraction ----------------
// grid = (BS/16, CH_), block = 256 (8 waves). Wave `wid` handles 8 i-values,
// computes 16x16 hyper tiles with v_wmma, folds them into xacc, ds_add_f32-reduces.
__global__ __launch_bounds__(256) void k_hyper(
    const unsigned short* __restrict__ h1bf, const unsigned short* __restrict__ w2bf,
    const float* __restrict__ xin, const float* __restrict__ b2,
    float* __restrict__ partials) {
  __shared__ float red[16 * H_];
  const int tid = threadIdx.x;
  const int lane = tid & 31, wid = tid >> 5;
  const int col = lane & 15, half = lane >> 4;
  const int b0 = blockIdx.x * 16;
  const int i0 = blockIdx.y * ICH_ + wid * IPW_;

  for (int t = tid; t < 16 * H_; t += 256) red[t] = 0.f;
  __syncthreads();

  // A = h1 tile (16 x 256), resident in VGPRs for all tiles of this wave
  const bf16_t* arow = (const bf16_t*)h1bf + (size_t)(b0 + col) * E_;
  v16bf afrag[8];
#pragma unroll
  for (int ks = 0; ks < 8; ++ks) afrag[ks] = load_frag(arow, ks * 32, half);

  // x_in scales for this wave's 8 i-values (C-layout row = r + 8*half)
  float xin_all[IPW_][8];
#pragma unroll
  for (int ii = 0; ii < IPW_; ++ii)
#pragma unroll
    for (int r = 0; r < 8; ++r)
      xin_all[ii][r] = xin[(size_t)(b0 + r + 8 * half) * IN_ + (i0 + ii)];

  for (int ht = 0; ht < 16; ++ht) {
    float xacc[8] = {0.f, 0.f, 0.f, 0.f, 0.f, 0.f, 0.f, 0.f};
#pragma unroll
    for (int ii = 0; ii < IPW_; ++ii) {
      const int n0 = (i0 + ii) * H_ + ht * 16;  // flat row of hy_w2
      const bf16_t* brow = (const bf16_t*)w2bf + (size_t)(n0 + col) * E_;
      __builtin_prefetch(brow + 16 * E_, 0, 1);  // next ht tile's row stream
      v8f acc = {};
#pragma unroll
      for (int ks = 0; ks < 8; ++ks) {
        v16bf bfrag = load_frag(brow, ks * 32, half);
        acc = __builtin_amdgcn_wmma_f32_16x16x32_bf16(
            false, afrag[ks], false, bfrag, (short)0, acc, false, false);
      }
      const float bv = b2[n0 + col];
#pragma unroll
      for (int r = 0; r < 8; ++r) {
        float w = fast_tanh(acc[r] + bv) * 3.0f;
        xacc[r] = fmaf(xin_all[ii][r], w, xacc[r]);
      }
    }
#pragma unroll
    for (int r = 0; r < 8; ++r)
      atomicAdd(&red[(r + 8 * half) * H_ + ht * 16 + col], xacc[r]);
  }
  __syncthreads();
  float* dst = partials + ((size_t)blockIdx.y * BS_ + b0) * H_;
  for (int t = tid; t < 16 * H_; t += 256) dst[t] = red[t];
}

// xpre = sum_c partials + bias row of hyper (computed in f32 here)
__global__ __launch_bounds__(256) void k_bias_reduce(
    const float* __restrict__ h1f, const float* __restrict__ w2,
    const float* __restrict__ b2, const float* __restrict__ partials,
    float* __restrict__ xpre) {
  int idx = blockIdx.x * 256 + threadIdx.x;  // b*H + h
  int b = idx >> 8, h = idx & 255;
  const float4* hr = (const float4*)(h1f + (size_t)b * E_);
  const float4* wr = (const float4*)(w2 + (size_t)(NW_ + h) * E_);
  float s = 0.f;
#pragma unroll 8
  for (int j = 0; j < E_ / 4; ++j) {
    float4 a = hr[j], w = wr[j];
    s += a.x * w.x + a.y * w.y + a.z * w.z + a.w * w.w;
  }
  float x = fast_tanh(s + b2[NW_ + h]) * 3.0f;
#pragma unroll
  for (int c = 0; c < CH_; ++c) x += partials[((size_t)c * BS_ + b) * H_ + h];
  xpre[idx] = x;
}

// ---------------- generic bf16 WMMA GEMM: C = A(MxK) @ Bw(NxK)^T + bias ----------------
// block = 128 (4 waves), one 16x16 tile per wave.
__global__ __launch_bounds__(128) void k_gemm_bf16(
    const unsigned short* __restrict__ A, const unsigned short* __restrict__ Bw,
    const float* __restrict__ bias, float* __restrict__ C,
    unsigned short* __restrict__ Cbf, int M, int N, int K) {
  const int lane = threadIdx.x & 31, wid = threadIdx.x >> 5;
  const int col = lane & 15, half = lane >> 4;
  const int ntile = N >> 4;
  int t = blockIdx.x * 4 + wid;
  int tm = t / ntile, tn = t - tm * ntile;
  int m0 = tm * 16, n0 = tn * 16;
  const bf16_t* arow = (const bf16_t*)A + (size_t)(m0 + col) * K;
  const bf16_t* brow = (const bf16_t*)Bw + (size_t)(n0 + col) * K;
  v8f acc = {};
  for (int k0 = 0; k0 < K; k0 += 32) {
    v16bf a = load_frag(arow, k0, half);
    v16bf b = load_frag(brow, k0, half);
    acc = __builtin_amdgcn_wmma_f32_16x16x32_bf16(
        false, a, false, b, (short)0, acc, false, false);
  }
  const float bv = bias ? bias[n0 + col] : 0.f;
#pragma unroll
  for (int r = 0; r < 8; ++r) {
    float v = acc[r] + bv;
    size_t o = (size_t)(m0 + r + 8 * half) * N + n0 + col;
    if (C) C[o] = v;
    if (Cbf) Cbf[o] = f2bf(v);
  }
}

// ---------------- GRU gate fusion + dueling head ----------------

__global__ __launch_bounds__(256) void k_gru(
    const float* __restrict__ gi, const float* __restrict__ gh,
    const float* __restrict__ hprev, float* __restrict__ hout) {
  int idx = blockIdx.x * 256 + threadIdx.x;  // b*H + h
  int b = idx >> 8, h = idx & 255;
  const float* gib = gi + (size_t)b * 3 * H_;
  const float* ghb = gh + (size_t)b * 3 * H_;
  float r = fast_sigmoid(gib[h] + ghb[h]);
  float z = fast_sigmoid(gib[H_ + h] + ghb[H_ + h]);
  float n = fast_tanh(gib[2 * H_ + h] + r * ghb[2 * H_ + h]);
  hout[idx] = (1.f - z) * n + z * hprev[idx];
}

__global__ __launch_bounds__(256) void k_head(
    const float* __restrict__ h, const float* __restrict__ val_w,
    const float* __restrict__ val_b, const float* __restrict__ adv_w,
    const float* __restrict__ adv_b, float* __restrict__ q) {
  int wid = threadIdx.x >> 5, lane = threadIdx.x & 31;
  int b = blockIdx.x * 8 + wid;
  const float4* hr = (const float4*)(h + (size_t)b * H_);
  float s = 0.f;
  const float* wrow = (lane < A_) ? (adv_w + (size_t)lane * H_)
                                  : ((lane == A_) ? val_w : nullptr);
  if (wrow) {
    const float4* wr = (const float4*)wrow;
#pragma unroll 8
    for (int j = 0; j < H_ / 4; ++j) {
      float4 a = hr[j], w = wr[j];
      s += a.x * w.x + a.y * w.y + a.z * w.z + a.w * w.w;
    }
    s += (lane < A_) ? adv_b[lane] : val_b[0];
  }
  float mean = wave_sum(lane < A_ ? s : 0.f) * (1.f / (float)A_);
  float val = __shfl(s, A_, 32);
  if (lane < A_) q[(size_t)b * A_ + lane] = val + s - mean;
}

// ---------------- host-side launch ----------------

extern "C" void kernel_launch(void* const* d_in, const int* in_sizes, int n_in,
                              void* d_out, int out_size, void* d_ws, size_t ws_size,
                              hipStream_t stream) {
  const float* inputs = (const float*)d_in[0];
  const float* hidden = (const float*)d_in[1];
  const float* goal   = (const float*)d_in[2];
  const float* ln1_g  = (const float*)d_in[3];
  const float* ln1_b  = (const float*)d_in[4];
  const float* hy_w1  = (const float*)d_in[5];
  const float* hy_b1  = (const float*)d_in[6];
  const float* hy_w2  = (const float*)d_in[7];
  const float* hy_b2  = (const float*)d_in[8];
  const float* ln2_g  = (const float*)d_in[9];
  const float* ln2_b  = (const float*)d_in[10];
  const float* fe_w   = (const float*)d_in[11];
  const float* fe_b   = (const float*)d_in[12];
  const float* w_ih   = (const float*)d_in[13];
  const float* w_hh   = (const float*)d_in[14];
  const float* b_ih   = (const float*)d_in[15];
  const float* b_hh   = (const float*)d_in[16];
  const float* val_w  = (const float*)d_in[17];
  const float* val_b  = (const float*)d_in[18];
  const float* adv_w  = (const float*)d_in[19];
  const float* adv_b  = (const float*)d_in[20];

  char* p = (char*)d_ws;
  auto take = [&](size_t bytes) -> void* {
    void* r = (void*)p;
    p += (bytes + 255) & ~(size_t)255;
    return r;
  };
  float*          xin      = (float*)take((size_t)BS_ * IN_ * 4);
  float*          h1f      = (float*)take((size_t)BS_ * E_ * 4);
  unsigned short* h1bf     = (unsigned short*)take((size_t)BS_ * E_ * 2);
  unsigned short* w2bf     = (unsigned short*)take((size_t)NW_ * E_ * 2);
  float*          partials = (float*)take((size_t)CH_ * BS_ * H_ * 4);
  float*          xpre     = (float*)take((size_t)BS_ * H_ * 4);
  unsigned short* x2bf     = (unsigned short*)take((size_t)BS_ * H_ * 2);
  unsigned short* x3bf     = (unsigned short*)take((size_t)BS_ * H_ * 2);
  float*          gi       = (float*)take((size_t)BS_ * 3 * H_ * 4);
  float*          gh       = (float*)take((size_t)BS_ * 3 * H_ * 4);
  unsigned short* febf     = (unsigned short*)take((size_t)H_ * H_ * 2);
  unsigned short* wihbf    = (unsigned short*)take((size_t)3 * H_ * H_ * 2);
  unsigned short* whhbf    = (unsigned short*)take((size_t)3 * H_ * H_ * 2);
  unsigned short* hsbf     = (unsigned short*)take((size_t)BS_ * H_ * 2);

  float* q_out = (float*)d_out;                     // (BS, 30)
  float* h_out = (float*)d_out + (size_t)BS_ * A_;  // (BS, 256)

  // 1. LN1 and goal MLP
  k_layernorm<<<BS_ / 8, 256, 0, stream>>>(inputs, ln1_g, ln1_b, xin, BS_);
  k_h1<<<BS_ * E_ / 256, 256, 0, stream>>>(goal, hy_w1, hy_b1, h1f, h1bf);

  // 2. bf16 conversions (one pass each; hy_w2 stays resident in L2 afterwards)
  k_f32_to_bf16<<<(NW_ * E_) / 256, 256, 0, stream>>>(hy_w2, w2bf, (size_t)NW_ * E_);
  k_f32_to_bf16<<<(H_ * H_) / 256, 256, 0, stream>>>(fe_w, febf, (size_t)H_ * H_);
  k_f32_to_bf16<<<(3 * H_ * H_) / 256, 256, 0, stream>>>(w_ih, wihbf, (size_t)3 * H_ * H_);
  k_f32_to_bf16<<<(3 * H_ * H_) / 256, 256, 0, stream>>>(w_hh, whhbf, (size_t)3 * H_ * H_);
  k_f32_to_bf16<<<(BS_ * H_) / 256, 256, 0, stream>>>(hidden, hsbf, (size_t)BS_ * H_);

  // 3. fused hypernet GEMM + tanh + per-sample contraction (the 138 GFLOP step)
  k_hyper<<<dim3(BS_ / 16, CH_), 256, 0, stream>>>(h1bf, w2bf, xin, hy_b2, partials);

  // 4. chunk reduce + hyper bias row
  k_bias_reduce<<<BS_ * H_ / 256, 256, 0, stream>>>(h1f, hy_w2, hy_b2, partials, xpre);

  // 5. LN2 + relu -> bf16
  k_ln_relu_bf<<<BS_ / 8, 256, 0, stream>>>(xpre, ln2_g, ln2_b, x2bf, BS_);

  // 6. fe GEMM (bf16 WMMA), 7. GRU gate GEMMs
  k_gemm_bf16<<<(BS_ / 16) * (H_ / 16) / 4, 128, 0, stream>>>(
      x2bf, febf, fe_b, nullptr, x3bf, BS_, H_, H_);
  k_gemm_bf16<<<(BS_ / 16) * (3 * H_ / 16) / 4, 128, 0, stream>>>(
      x3bf, wihbf, b_ih, gi, nullptr, BS_, 3 * H_, H_);
  k_gemm_bf16<<<(BS_ / 16) * (3 * H_ / 16) / 4, 128, 0, stream>>>(
      hsbf, whhbf, b_hh, gh, nullptr, BS_, 3 * H_, H_);

  // 8. GRU gating -> h, 9. dueling head -> q
  k_gru<<<BS_ * H_ / 256, 256, 0, stream>>>(gi, gh, hidden, h_out);
  k_head<<<BS_ / 8, 256, 0, stream>>>(h_out, val_w, val_b, adv_w, adv_b, q_out);
}
